// MESH_NET_59304908423641
// MI455X (gfx1250) — compile-verified, matrix operand order checked by reading, not run on hardware
//
#include <hip/hip_runtime.h>
#include <math.h>

#define G_ 32
#define N_ 16384
#define E_ (G_ * N_ * 8)     // 4194304
#define K1_ 9831
#define K2_ 5899
#define K3_ 3540

typedef __attribute__((ext_vector_type(16))) _Float16 v16h;
typedef __attribute__((ext_vector_type(8)))  float    v8f;

__device__ __forceinline__ unsigned keyOf(float s) {
  unsigned u = __float_as_uint(s);
  return (u & 0x80000000u) ? ~u : (u | 0x80000000u);
}

// ---------------- utility fills / copies ----------------
__global__ void k_fill_f32(float* p, float v, int n) {
  int i = blockIdx.x * blockDim.x + threadIdx.x;
  if (i < n) p[i] = v;
}
__global__ void k_fill_i32(int* p, int v, int n) {
  int i = blockIdx.x * blockDim.x + threadIdx.x;
  if (i < n) p[i] = v;
}
__global__ void k_init_edges(const int* __restrict__ ei, const float* __restrict__ ew,
                             int* rw, int* cw, float* eww) {
  int i = blockIdx.x * blockDim.x + threadIdx.x;
  if (i < E_) { rw[i] = ei[i]; cw[i] = ei[E_ + i]; eww[i] = ew[i]; }
}
__global__ void k_pad_w(const float* __restrict__ W, float* Wp, int kdim) {
  int i = blockIdx.x * blockDim.x + threadIdx.x;
  if (i < 1024) Wp[i] = (i < kdim * 32) ? W[i] : 0.0f;
}

// ---------------- WMMA GEMM: xw[n,32] = x[n,kdim] @ W[32,32] ----------------
// One wave per 16 rows; 2 x v_wmma_f32_16x16x32_f16 per wave (N tiles 0..15, 16..31).
__global__ void __launch_bounds__(256) k_gemm_wmma(
    const float* __restrict__ x, int ldx, int kdim,
    const float* __restrict__ W, float* __restrict__ xw, int nRows) {
  int wave = (blockIdx.x * blockDim.x + threadIdx.x) >> 5;
  int lane = threadIdx.x & 31;
  int m0 = wave << 4;
  if (m0 >= nRows) return;               // wave-uniform: EXEC stays all-ones
  int mloc = lane & 15;
  int half = lane >> 4;
  int node = m0 + mloc;
  const float* xr = x + (size_t)node * ldx;
  v16h a;
#pragma unroll
  for (int e = 0; e < 16; ++e) {
    int k = ((e < 8) ? 0 : 16) + half * 8 + (e & 7);
    float v = (k < kdim) ? xr[k] : 0.0f;
    a[e] = (_Float16)v;
  }
  v16h b0, b1;
#pragma unroll
  for (int e = 0; e < 16; ++e) {
    int k = ((e < 8) ? 0 : 16) + half * 8 + (e & 7);
    b0[e] = (_Float16)W[k * 32 + mloc];
    b1[e] = (_Float16)W[k * 32 + mloc + 16];
  }
  v8f c0 = {}; v8f c1 = {};
  c0 = __builtin_amdgcn_wmma_f32_16x16x32_f16(false, a, false, b0, (short)0, c0, false, false);
  c1 = __builtin_amdgcn_wmma_f32_16x16x32_f16(false, a, false, b1, (short)0, c1, false, false);
#pragma unroll
  for (int r = 0; r < 8; ++r) {
    int m = m0 + r + half * 8;
    xw[(size_t)m * 32 + mloc]      = c0[r];
    xw[(size_t)m * 32 + mloc + 16] = c1[r];
  }
}

// ---------------- GCN message passing ----------------
__global__ void k_deg(const int* __restrict__ cw, const float* __restrict__ eww, float* deg) {
  int i = blockIdx.x * blockDim.x + threadIdx.x;
  if (i < E_) atomicAdd(&deg[cw[i]], eww[i]);
}
__global__ void k_rsqrt(float* d, int n) {
  int i = blockIdx.x * blockDim.x + threadIdx.x;
  if (i < n) d[i] = rsqrtf(d[i]);
}
__global__ void k_msg(const float* __restrict__ xw, const int* __restrict__ rw,
                      const int* __restrict__ cw, const float* __restrict__ eww,
                      const float* __restrict__ dis, float* agg) {
  long long t = (long long)blockIdx.x * blockDim.x + threadIdx.x;
  if (t >= (long long)E_ * 8) return;
  int e = (int)(t >> 3), q = (int)(t & 7);
  int r = rw[e], c = cw[e];
  float coef = dis[r] * eww[e] * dis[c];
  if (coef != 0.0f) {
    float4 v = ((const float4*)xw)[(size_t)r * 8 + q];
    float* ap = agg + (size_t)c * 32 + q * 4;
    atomicAdd(ap + 0, v.x * coef); atomicAdd(ap + 1, v.y * coef);
    atomicAdd(ap + 2, v.z * coef); atomicAdd(ap + 3, v.w * coef);
  }
}
__global__ void k_combine(const float* __restrict__ agg, const float* __restrict__ xw,
                          const float* __restrict__ dis, const float* __restrict__ b,
                          float* out, int n) {
  int i = blockIdx.x * blockDim.x + threadIdx.x;
  if (i < n * 32) {
    int node = i >> 5, j = i & 31;
    float d = dis[node];
    float v = agg[i] + xw[i] * d * d + b[j];
    out[i] = v > 0.0f ? v : 0.0f;
  }
}

// ---------------- TopK pooling ----------------
__global__ void k_score(const float* __restrict__ x, const float* __restrict__ pw,
                        float* sc, int n) {
  int i = blockIdx.x * blockDim.x + threadIdx.x;
  if (i >= n) return;
  float nrm = 0.f, dot = 0.f;
#pragma unroll
  for (int j = 0; j < 32; ++j) {
    float w = pw[j];
    nrm += w * w;
    dot += x[(size_t)i * 32 + j] * w;
  }
  sc[i] = dot * rsqrtf(nrm);
}

// 4-pass 8-bit radix select of the k-th largest score per graph (one block/graph).
__global__ void k_select(const float* __restrict__ sc, int n, int k,
                         unsigned* selT, int* selTies) {
  __shared__ unsigned hist[256];
  __shared__ unsigned s_prefix;
  __shared__ int s_kk;
  int g = blockIdx.x, tid = threadIdx.x;
  const float* s = sc + (size_t)g * n;
  if (tid == 0) { s_prefix = 0u; s_kk = k; }
  __syncthreads();
  for (int pass = 0; pass < 4; ++pass) {
    int shift = 24 - pass * 8;
    hist[tid] = 0u;
    __syncthreads();
    unsigned prefix = s_prefix;
    for (int i = tid; i < n; i += 256) {
      unsigned key = keyOf(s[i]);
      if (pass == 0 || ((key ^ prefix) >> (shift + 8)) == 0)
        atomicAdd(&hist[(key >> shift) & 255u], 1u);
    }
    __syncthreads();
    if (tid == 0) {
      int kk = s_kk;
      unsigned bsel = 0;
      for (int bb = 255; bb >= 0; --bb) {
        unsigned h = hist[bb];
        if ((unsigned)kk <= h) { bsel = (unsigned)bb; break; }
        kk -= (int)h;
      }
      s_kk = kk;
      s_prefix = prefix | (bsel << shift);
    }
    __syncthreads();
  }
  if (tid == 0) { selT[g] = s_prefix; selTies[g] = s_kk; }
}

// Deterministic stable compaction: assign new ids 0..k-1 per graph.
__global__ void k_compact(const float* __restrict__ sc, int n, int k,
                          const unsigned* __restrict__ selT, const int* __restrict__ selTies,
                          int* old2new, int* kept) {
  __shared__ unsigned scn[256];
  __shared__ unsigned sBaseG, sBaseE;
  int g = blockIdx.x, tid = threadIdx.x;
  unsigned T = selT[g];
  int t = selTies[g];
  int kG = k - t;                          // number of strictly-greater slots
  if (tid == 0) { sBaseG = 0u; sBaseE = 0u; }
  __syncthreads();
  for (int base = 0; base < n; base += 256) {
    int il = base + tid;
    bool valid = il < n;
    unsigned key = valid ? keyOf(sc[(size_t)g * n + il]) : 0u;
    unsigned gf = (valid && key > T) ? 1u : 0u;
    unsigned ef = (valid && key == T) ? 1u : 0u;
    unsigned p = gf | (ef << 16);
    scn[tid] = p;
    __syncthreads();
    for (int off = 1; off < 256; off <<= 1) {
      unsigned v = (tid >= off) ? scn[tid - off] : 0u;
      __syncthreads();
      scn[tid] += v;
      __syncthreads();
    }
    unsigned excl = scn[tid] - p;
    unsigned tot = scn[255];
    unsigned bG = sBaseG, bE = sBaseE;
    int newl = -1;
    if (gf) newl = (int)(bG + (excl & 0xFFFFu));
    else if (ef) {
      unsigned re = bE + (excl >> 16);
      if ((int)re < t) newl = kG + (int)re;
    }
    if (newl >= 0) {
      int old = g * n + il;
      int ng = g * k + newl;
      old2new[old] = ng;
      kept[ng] = old;
    }
    __syncthreads();
    if (tid == 0) { sBaseG += tot & 0xFFFFu; sBaseE += tot >> 16; }
    __syncthreads();
  }
}

__global__ void k_gather(const float* __restrict__ x, const float* __restrict__ sc,
                         const int* __restrict__ kept, float* xn, int nNew) {
  int i = blockIdx.x * blockDim.x + threadIdx.x;
  if (i >= nNew * 32) return;
  int ni = i >> 5, j = i & 31;
  int old = kept[ni];
  xn[i] = x[(size_t)old * 32 + j] * tanhf(sc[old]);
}

__global__ void k_remap(int* rw, int* cw, float* eww, const int* __restrict__ old2new, int mode) {
  int i = blockIdx.x * blockDim.x + threadIdx.x;
  if (i >= E_) return;
  bool pv = (mode == 0) ? true : (eww[i] > 0.5f);
  int r = old2new[rw[i]], c = old2new[cw[i]];
  bool keep = pv && r >= 0 && c >= 0;
  rw[i] = keep ? r : 0;
  cw[i] = keep ? c : 0;
  eww[i] = keep ? 1.0f : 0.0f;
}

// readout: [G,64] = concat(max over k, mean over k); accumulated into hsum
__global__ void k_readout(const float* __restrict__ x, int k, float* hsum) {
  __shared__ float smx[256], ssm[256];
  int g = blockIdx.x, tid = threadIdx.x, rg = tid >> 5, f = tid & 31;
  float mx = -1e30f, sm = 0.f;
  for (int i = rg; i < k; i += 8) {
    float v = x[((size_t)g * k + i) * 32 + f];
    mx = fmaxf(mx, v);
    sm += v;
  }
  smx[tid] = mx; ssm[tid] = sm;
  __syncthreads();
  for (int s = 128; s >= 32; s >>= 1) {
    if (tid < s) { smx[tid] = fmaxf(smx[tid], smx[tid + s]); ssm[tid] += ssm[tid + s]; }
    __syncthreads();
  }
  if (tid < 32) {
    hsum[g * 64 + tid] += smx[tid];
    hsum[g * 64 + 32 + tid] += ssm[tid] / (float)k;
  }
}

__global__ void k_mlp(const float* __restrict__ hsum,
                      const float* __restrict__ W1, const float* __restrict__ b1,
                      const float* __restrict__ W2, const float* __restrict__ b2,
                      float* out) {
  int tid = threadIdx.x;
  int gg = tid >> 3, o = tid & 7;
  float hid[16];
#pragma unroll
  for (int q = 0; q < 16; ++q) {
    float acc = b1[q];
    for (int u = 0; u < 64; ++u) acc += hsum[gg * 64 + u] * W1[u * 16 + q];
    hid[q] = acc > 0.f ? acc : 0.f;
  }
  float acc = b2[o];
#pragma unroll
  for (int q = 0; q < 16; ++q) acc += hid[q] * W2[q * 8 + o];
  out[gg * 8 + o] = 1.0f / (1.0f + expf(-acc));
}

static inline int cdiv(long long a, long long b) { return (int)((a + b - 1) / b); }

extern "C" void kernel_launch(void* const* d_in, const int* in_sizes, int n_in,
                              void* d_out, int out_size, void* d_ws, size_t ws_size,
                              hipStream_t stream) {
  const float* pos = (const float*)d_in[0];
  const int*   ei  = (const int*)d_in[1];
  const float* ew  = (const float*)d_in[2];
  const float* W1  = (const float*)d_in[4];  const float* b1 = (const float*)d_in[5];
  const float* W2  = (const float*)d_in[6];  const float* b2 = (const float*)d_in[7];
  const float* W3  = (const float*)d_in[8];  const float* b3 = (const float*)d_in[9];
  const float* pw1 = (const float*)d_in[10];
  const float* pw2 = (const float*)d_in[11];
  const float* pw3 = (const float*)d_in[12];
  const float* l1W = (const float*)d_in[13]; const float* l1b = (const float*)d_in[14];
  const float* l2W = (const float*)d_in[15]; const float* l2b = (const float*)d_in[16];
  float* out = (float*)d_out;

  const int n0 = G_ * N_, n1 = G_ * K1_, n2 = G_ * K2_;
  char* base = (char*)d_ws;
  size_t off = 0;
  auto alloc = [&](size_t bytes) { void* p = base + off; off += (bytes + 255) & ~(size_t)255; return p; };
  float* bufA = (float*)alloc((size_t)n0 * 32 * 4);
  float* bufB = (float*)alloc((size_t)n0 * 32 * 4);
  float* bufC = (float*)alloc((size_t)n0 * 32 * 4);
  int*   rw   = (int*)alloc((size_t)E_ * 4);
  int*   cw   = (int*)alloc((size_t)E_ * 4);
  float* eww  = (float*)alloc((size_t)E_ * 4);
  float* dis  = (float*)alloc((size_t)n0 * 4);
  float* sc   = (float*)alloc((size_t)n0 * 4);
  int*   o2n  = (int*)alloc((size_t)n0 * 4);
  int*   kept = (int*)alloc((size_t)n0 * 4);
  unsigned* selT = (unsigned*)alloc(G_ * 4);
  int*   selTies = (int*)alloc(G_ * 4);
  float* hsum = (float*)alloc(G_ * 64 * 4);
  float* wpad = (float*)alloc(32 * 32 * 4);

  k_init_edges<<<cdiv(E_, 256), 256, 0, stream>>>(ei, ew, rw, cw, eww);
  k_fill_f32<<<cdiv(G_ * 64, 256), 256, 0, stream>>>(hsum, 0.f, G_ * 64);
  k_pad_w<<<4, 256, 0, stream>>>(W1, wpad, 3);

  auto run_gcn = [&](const float* xin, int ldx, int kdim, const float* W, const float* bias,
                     int n, float* xw, float* agg_x) {
    k_fill_f32<<<cdiv((long long)n * 32, 256), 256, 0, stream>>>(agg_x, 0.f, n * 32);
    k_fill_f32<<<cdiv(n, 256), 256, 0, stream>>>(dis, 1.0f, n);   // self-loop: deg starts at 1
    int waves = n / 16;                                            // n is a multiple of 16
    k_gemm_wmma<<<cdiv(waves, 8), 256, 0, stream>>>(xin, ldx, kdim, W, xw, n);
    k_deg<<<cdiv(E_, 256), 256, 0, stream>>>(cw, eww, dis);
    k_rsqrt<<<cdiv(n, 256), 256, 0, stream>>>(dis, n);
    k_msg<<<cdiv((long long)E_ * 8, 256), 256, 0, stream>>>(xw, rw, cw, eww, dis, agg_x);
    k_combine<<<cdiv((long long)n * 32, 256), 256, 0, stream>>>(agg_x, xw, dis, bias, agg_x, n);
  };
  auto run_pool = [&](const float* xin, const float* pw, int nper, int kper, int n, int mode,
                      float* xn) {
    k_score<<<cdiv(n, 256), 256, 0, stream>>>(xin, pw, sc, n);
    k_select<<<G_, 256, 0, stream>>>(sc, nper, kper, selT, selTies);
    k_fill_i32<<<cdiv(n, 256), 256, 0, stream>>>(o2n, -1, n);
    k_compact<<<G_, 256, 0, stream>>>(sc, nper, kper, selT, selTies, o2n, kept);
    int nNew = G_ * kper;
    k_gather<<<cdiv((long long)nNew * 32, 256), 256, 0, stream>>>(xin, sc, kept, xn, nNew);
    k_remap<<<cdiv(E_, 256), 256, 0, stream>>>(rw, cw, eww, o2n, mode);
    k_readout<<<G_, 256, 0, stream>>>(xn, kper, hsum);
  };

  // layer 1: conv1 (pos 3->32), pool1, readout1
  run_gcn(pos, 3, 3, wpad, b1, n0, bufB, bufC);
  run_pool(bufC, pw1, N_, K1_, n0, 0, bufA);
  // layer 2
  run_gcn(bufA, 32, 32, W2, b2, n1, bufB, bufC);
  run_pool(bufC, pw2, K1_, K2_, n1, 1, bufA);
  // layer 3
  run_gcn(bufA, 32, 32, W3, b3, n2, bufB, bufC);
  run_pool(bufC, pw3, K2_, K3_, n2, 1, bufA);

  k_mlp<<<1, 256, 0, stream>>>(hsum, l1W, l1b, l2W, l2b, out);
  (void)in_sizes; (void)n_in; (void)out_size; (void)ws_size;
}
